// MultiHeadAttention_21174188769772
// MI455X (gfx1250) — matrix operand order repaired and checked by reading, not compiled, not run on hardware
//
#include <hip/hip_runtime.h>

// ---------------------------------------------------------------------------
// MI455X / gfx1250 multi-head attention forward.
// Matmuls on v_wmma_f32_16x16x32_bf16; bf16 tile staging via the Tensor Data
// Mover (tensor_load_to_lds + s_wait_tensorcnt) with double-buffered LDS so
// the DMA of tile t+1 overlaps the WMMA work on tile t (1 barrier / k-step).
// fp32 sources converted in-flight with v_cvt_pk_bf16_f32.
// ---------------------------------------------------------------------------

typedef __attribute__((ext_vector_type(16))) __bf16 v16bf;
typedef __attribute__((ext_vector_type(8)))  float  v8f;
typedef __attribute__((ext_vector_type(4)))  unsigned int v4u;
typedef __attribute__((ext_vector_type(8)))  int    v8i;
typedef __attribute__((ext_vector_type(4)))  int    v4i;

#define TILE_K     32
#define LDS_STRIDE 40               // 32 K-elements + 8 pad (80B pitch)
#define BUF_ELEMS  (128 * LDS_STRIDE)

union FragAB {
    v16bf  v;
    uint4  q[2];
    __bf16 e[16];
};

__device__ __forceinline__ __bf16 f2bf(float f) { return (__bf16)f; }

// ---------------------------------------------------------------------------
// TDM: DMA a (rows x 32) bf16 tile, row pitch strideElems, from global memory
// into LDS at lds_byte, inserting 16B of LDS pad after every 64B row so the
// LDS image has the 80B pitch the WMMA fragment loaders expect.
// D# layout per CDNA5 ISA ch.8. Issued once per wave (EXEC ignored).
// ---------------------------------------------------------------------------
__device__ __forceinline__ void tdm_load_tile_bf16(const void* gsrc,
                                                   unsigned lds_byte,
                                                   unsigned rows,
                                                   long long strideElems)
{
    unsigned long long ga = (unsigned long long)(uintptr_t)gsrc;
    v4u g0;
    g0[0] = 1u;                                                // count = 1
    g0[1] = lds_byte;                                          // lds_addr
    g0[2] = (unsigned)ga;                                      // global_addr[31:0]
    g0[3] = (unsigned)((ga >> 32) & 0x01FFFFFFu) | (2u << 30); // addr[56:32], type=2

    const unsigned long long s0 = (unsigned long long)strideElems;
    const unsigned td0 = 1u << 24;   // tensor_dim0 OOB bound (never exceeded)
    const unsigned td1 = 1u << 20;   // tensor_dim1 OOB bound
    v8i g1;
    g1[0] = (int)((1u << 16) |       // data_size = 1 -> 2 bytes
                  (1u << 20) |       // pad_enable
                  (3u << 22) |       // pad_interval = 16 DWORDs (64B)
                  (3u << 25));       // pad_amount   = 4 DWORDs (16B)
    g1[1] = (int)((td0 & 0xFFFFu) << 16);                      // tensor_dim0[15:0]
    g1[2] = (int)((td0 >> 16) | ((td1 & 0xFFFFu) << 16));      // td0 hi | td1 lo
    g1[3] = (int)((td1 >> 16) | (32u << 16));                  // td1 hi | tile_dim0=32
    g1[4] = (int)(rows & 0xFFFFu);                             // tile_dim1 | tile_dim2=0
    g1[5] = (int)(unsigned)(s0 & 0xFFFFFFFFu);                 // dim0 stride lo
    g1[6] = (int)(unsigned)((s0 >> 32) & 0xFFFFu);             // stride hi | dim1 str=0
    g1[7] = 0;

    v4i z4 = {0, 0, 0, 0};                    // 2D tensor: groups 2/3 unused
    v8i z8 = {0, 0, 0, 0, 0, 0, 0, 0};
    __builtin_amdgcn_tensor_load_to_lds(g0, g1, z4, z4, z8, 0);
}

// ---------------------------------------------------------------------------
// Fold Wl [D, D*H] -> Wl_eff [D, D] (bf16):  Wl_eff[i,j] = sum_h Wl[i, h*D+j]
// ---------------------------------------------------------------------------
__global__ __launch_bounds__(256) void fold_wl_kernel(const float* __restrict__ Wl,
                                                      __bf16* __restrict__ Weff)
{
    const int idx = blockIdx.x * 256 + threadIdx.x;   // 0 .. D*D-1
    const int i = idx >> 10;                          // D = 1024
    const int j = idx & 1023;
    float s = 0.0f;
#pragma unroll
    for (int h = 0; h < 8; ++h)
        s += Wl[(size_t)i * 8192 + (size_t)h * 1024 + j];
    Weff[idx] = f2bf(s);
}

// ---------------------------------------------------------------------------
// Generic WMMA GEMM:   C[m,n] = scale * sum_k A[m,k] * B(k,n)  (+ bias[n])
//   BTRANS=1 : B(k,n) = Bsrc[n*ldb + k]
//   BTRANS=0 : B(k,n) = Bsrc[k*ldb + n]
// Block = 256 threads (8 waves). Block tile 128x128, wave tile 64x32
// (waves 2x4), sub-tiles 4x2 of 16x16.  K step = 32.
// Software pipeline: stage tile t+1 into LDS buffer (t+1)&1 (TDM for bf16
// sources, manual convert/transpose otherwise) while computing tile t.
// ---------------------------------------------------------------------------
template <bool AF32, bool BF32, bool BTRANS, bool CF32, bool HASBIAS>
__global__ __launch_bounds__(256)
void gemm_wmma_kernel(const void* __restrict__ Av, long long batchA, int lda,
                      const void* __restrict__ Bv, long long batchB, int ldb,
                      const float* __restrict__ bias,
                      void* __restrict__ Cv, long long batchC, int ldc,
                      int K, float scale)
{
    constexpr bool TDM_A = !AF32;               // bf16 A tile -> TDM
    constexpr bool TDM_B = (BTRANS && !BF32);   // bf16 transposed-weight B -> TDM

    __shared__ __align__(16) __bf16 As[2 * BUF_ELEMS];
    __shared__ __align__(16) __bf16 Bs[2 * BUF_ELEMS];

    const int tid  = threadIdx.x;
    const int lane = tid & 31;
    const int wid  = tid >> 5;      // 0..7
    const int wm   = wid >> 2;      // 0..1  (wave row)
    const int wn   = wid & 3;       // 0..3  (wave col)
    const int lrow = lane & 15;
    const int lhi  = lane >> 4;

    const int nblk = blockIdx.x * 128;
    const int mblk = blockIdx.y * 128;
    const int z    = blockIdx.z;

    const float*  Af32 = (const float*) Av + (size_t)z * (size_t)batchA;
    const __bf16* Abf  = (const __bf16*)Av + (size_t)z * (size_t)batchA;
    const float*  Bf32 = (const float*) Bv + (size_t)z * (size_t)batchB;
    const __bf16* Bbf  = (const __bf16*)Bv + (size_t)z * (size_t)batchB;

    v8f acc[4][2];
#pragma unroll
    for (int i = 0; i < 4; ++i)
#pragma unroll
        for (int j = 0; j < 2; ++j)
            acc[i][j] = 0.0f;

    // manual-staging thread mapping (contiguous 16-element chunks)
    const int srow = tid >> 1;          // 0..127
    const int sseg = (tid & 1) * 16;    // 0 or 16

    // Stage tiles for k-offset kb into LDS buffers Asb / Bsb.
    auto stage = [&](int kb, __bf16* Asb, __bf16* Bsb) {
        // ---- A tile: Asb[row][k]
        if (TDM_A) {
            if (wid == 0)
                tdm_load_tile_bf16(Abf + (size_t)mblk * lda + kb,
                                   (unsigned)(uintptr_t)Asb, 128, lda);
        } else {
            __bf16 tmp[16];
            const float4* gp =
                (const float4*)(Af32 + (size_t)(mblk + srow) * lda + kb + sseg);
#pragma unroll
            for (int i = 0; i < 4; ++i) {
                float4 f = gp[i];
                tmp[4 * i + 0] = f2bf(f.x);
                tmp[4 * i + 1] = f2bf(f.y);
                tmp[4 * i + 2] = f2bf(f.z);
                tmp[4 * i + 3] = f2bf(f.w);
            }
            uint4* sp = (uint4*)(Asb + srow * LDS_STRIDE + sseg);
            sp[0] = ((uint4*)tmp)[0];
            sp[1] = ((uint4*)tmp)[1];
            if (kb + TILE_K < K)
                __builtin_prefetch(Af32 + (size_t)(mblk + srow) * lda + kb + TILE_K + sseg);
        }
        // ---- B tile: Bsb[n][k]
        if (BTRANS) {
            if (TDM_B) {
                if (wid == 1)
                    tdm_load_tile_bf16(Bbf + (size_t)nblk * ldb + kb,
                                       (unsigned)(uintptr_t)Bsb, 128, ldb);
            } else {
                __bf16 tmp[16];
                const float4* gp =
                    (const float4*)(Bf32 + (size_t)(nblk + srow) * ldb + kb + sseg);
#pragma unroll
                for (int i = 0; i < 4; ++i) {
                    float4 f = gp[i];
                    tmp[4 * i + 0] = f2bf(f.x);
                    tmp[4 * i + 1] = f2bf(f.y);
                    tmp[4 * i + 2] = f2bf(f.z);
                    tmp[4 * i + 3] = f2bf(f.w);
                }
                uint4* sp = (uint4*)(Bsb + srow * LDS_STRIDE + sseg);
                sp[0] = ((uint4*)tmp)[0];
                sp[1] = ((uint4*)tmp)[1];
                if (kb + TILE_K < K)
                    __builtin_prefetch(Bf32 + (size_t)(nblk + srow) * ldb + kb + TILE_K + sseg);
            }
        } else {
            // direct row-major B: transpose into Bsb[n][k]
            const int k  = tid >> 3;          // 0..31
            const int n0 = (tid & 7) * 16;    // 0..112
            __bf16 tmp[16];
            const uint4* gp =
                (const uint4*)(Bbf + (size_t)(kb + k) * ldb + nblk + n0);
            ((uint4*)tmp)[0] = gp[0];
            ((uint4*)tmp)[1] = gp[1];
#pragma unroll
            for (int i = 0; i < 16; ++i)
                Bsb[(n0 + i) * LDS_STRIDE + k] = tmp[i];
            if (kb + TILE_K < K)
                __builtin_prefetch(Bbf + (size_t)(kb + TILE_K + k) * ldb + nblk + n0);
        }
    };

    const int ntiles = K / TILE_K;

    // prologue: stage tile 0 into buffer 0
    stage(0, As, Bs);

    for (int t = 0; t < ntiles; ++t) {
        // Publish tile t: issuing waves drain TENSORcnt; the barrier also
        // drains DScnt (manual staging) and guarantees everyone is done
        // reading buffer (t+1)&1 from iteration t-1 before it is re-filled.
        if (TDM_A || TDM_B) {
            if (wid <= 1) __builtin_amdgcn_s_wait_tensorcnt(0);
        }
        __syncthreads();

        // stage tile t+1 into the other buffer (overlaps with compute below)
        if (t + 1 < ntiles)
            stage((t + 1) * TILE_K, As + ((t + 1) & 1) * BUF_ELEMS,
                                    Bs + ((t + 1) & 1) * BUF_ELEMS);

        const __bf16* Acur = As + (t & 1) * BUF_ELEMS;
        const __bf16* Bcur = Bs + (t & 1) * BUF_ELEMS;

        // ------------------- WMMA compute on tile t ---------------------
        // A frag (16x32 bf16): lane=row, VGPR pairs at K = lhi*8 + {0..7,16..23}
        // -> two ds_load_b128 at +0 / +16 elems.
        // B frag (32x16 bf16): lane=col, VGPR pairs at K = lhi*8 + {0..7,16..23}.
        FragAB a[4], b[2];
#pragma unroll
        for (int sm = 0; sm < 4; ++sm) {
            const __bf16* p = Acur + (wm * 64 + sm * 16 + lrow) * LDS_STRIDE + lhi * 8;
            a[sm].q[0] = *(const uint4*)(p);
            a[sm].q[1] = *(const uint4*)(p + 16);
        }
#pragma unroll
        for (int sn = 0; sn < 2; ++sn) {
            const __bf16* p = Bcur + (wn * 32 + sn * 16 + lrow) * LDS_STRIDE + lhi * 8;
            b[sn].q[0] = *(const uint4*)(p);
            b[sn].q[1] = *(const uint4*)(p + 16);
        }
#pragma unroll
        for (int sm = 0; sm < 4; ++sm)
#pragma unroll
            for (int sn = 0; sn < 2; ++sn)
                acc[sm][sn] = __builtin_amdgcn_wmma_f32_16x16x32_bf16(
                    false, a[sm].v, false, b[sn].v,
                    (short)0, acc[sm][sn], false, false);
    }

    // ------------------- epilogue -----------------------------------------
    // C/D layout: VGPR r, lane L -> row (r + 8*(L>>4)), col (L&15).
    float*  Cf = (float*) Cv + (size_t)z * (size_t)batchC;
    __bf16* Cb = (__bf16*)Cv + (size_t)z * (size_t)batchC;
#pragma unroll
    for (int sm = 0; sm < 4; ++sm) {
        const int m0 = mblk + wm * 64 + sm * 16 + lhi * 8;
#pragma unroll
        for (int sn = 0; sn < 2; ++sn) {
            const int n  = nblk + wn * 32 + sn * 16 + lrow;
            const float bv = HASBIAS ? bias[n] : 0.0f;
#pragma unroll
            for (int r = 0; r < 8; ++r) {
                const float val = acc[sm][sn][r] * scale + bv;
                if (CF32) Cf[(size_t)(m0 + r) * ldc + n] = val;
                else      Cb[(size_t)(m0 + r) * ldc + n] = f2bf(val);
            }
        }
    }
}

// ---------------------------------------------------------------------------
// Row softmax, in place over bf16 scores [B*S rows, S=2048 cols].
// ---------------------------------------------------------------------------
__global__ __launch_bounds__(256) void softmax_kernel(__bf16* __restrict__ scores)
{
    __shared__ float red[256];
    const int row = blockIdx.x;
    const int tid = threadIdx.x;
    __bf16* rp = scores + (size_t)row * 2048;

    float x[8];
    float mx = -3.4e38f;
#pragma unroll
    for (int i = 0; i < 8; ++i) {
        x[i] = (float)rp[tid + i * 256];
        mx   = fmaxf(mx, x[i]);
    }
    red[tid] = mx;
    __syncthreads();
    for (int s = 128; s > 0; s >>= 1) {
        if (tid < s) red[tid] = fmaxf(red[tid], red[tid + s]);
        __syncthreads();
    }
    mx = red[0];
    __syncthreads();

    float sum = 0.0f;
#pragma unroll
    for (int i = 0; i < 8; ++i) {
        x[i] = __expf(x[i] - mx);
        sum += x[i];
    }
    red[tid] = sum;
    __syncthreads();
    for (int s = 128; s > 0; s >>= 1) {
        if (tid < s) red[tid] += red[tid + s];
        __syncthreads();
    }
    const float inv = 1.0f / red[0];
#pragma unroll
    for (int i = 0; i < 8; ++i)
        rp[tid + i * 256] = f2bf(x[i] * inv);
}

// ---------------------------------------------------------------------------
extern "C" void kernel_launch(void* const* d_in, const int* in_sizes, int n_in,
                              void* d_out, int out_size, void* d_ws, size_t ws_size,
                              hipStream_t stream)
{
    (void)in_sizes; (void)n_in; (void)out_size; (void)ws_size;

    const float* Q  = (const float*)d_in[0];
    const float* Km = (const float*)d_in[1];
    const float* V  = (const float*)d_in[2];
    const float* WQ = (const float*)d_in[3];
    const float* bQ = (const float*)d_in[4];
    const float* WK = (const float*)d_in[5];
    const float* bK = (const float*)d_in[6];
    const float* WV = (const float*)d_in[7];
    const float* bV = (const float*)d_in[8];
    const float* Wl = (const float*)d_in[9];
    const float* bl = (const float*)d_in[10];
    float* out = (float*)d_out;

    const int B = 4, S = 2048, D = 1024;
    const size_t qkvBytes = (size_t)B * S * D * 2;   // 16 MB (bf16)

    char* ws = (char*)d_ws;
    __bf16* qb   = (__bf16*)(ws + 0 * qkvBytes);
    __bf16* kb   = (__bf16*)(ws + 1 * qkvBytes);
    __bf16* vb   = (__bf16*)(ws + 2 * qkvBytes);
    __bf16* Zb   = (__bf16*)(ws + 3 * qkvBytes);
    __bf16* Weff = (__bf16*)(ws + 4 * qkvBytes);                       // 2 MB
    __bf16* attn = (__bf16*)(ws + 4 * qkvBytes + (size_t)D * D * 2);   // 33.5 MB

    const dim3 blk(256);

    // 0) Wl_eff = sum_h Wl[:, h*D:(h+1)*D]   (8x FLOP reduction on out-proj)
    fold_wl_kernel<<<dim3((D * D) / 256), blk, 0, stream>>>(Wl, Weff);

    // 1) projections: q/k/v = X @ W.T + b   (M=8192, N=1024, K=1024), fp32 src
    gemm_wmma_kernel<true, true, true, false, true>
        <<<dim3(D / 128, (B * S) / 128, 1), blk, 0, stream>>>(
            Q, 0, D, WQ, 0, D, bQ, qb, 0, D, D, 1.0f);
    gemm_wmma_kernel<true, true, true, false, true>
        <<<dim3(D / 128, (B * S) / 128, 1), blk, 0, stream>>>(
            Km, 0, D, WK, 0, D, bK, kb, 0, D, D, 1.0f);
    gemm_wmma_kernel<true, true, true, false, true>
        <<<dim3(D / 128, (B * S) / 128, 1), blk, 0, stream>>>(
            V, 0, D, WV, 0, D, bV, vb, 0, D, D, 1.0f);

    // 2) scores = (q @ k^T) / 32  (batched, M=N=2048, K=1024), A+B via TDM
    gemm_wmma_kernel<false, false, true, false, false>
        <<<dim3(S / 128, S / 128, B), blk, 0, stream>>>(
            qb, (long long)S * D, D, kb, (long long)S * D, D, nullptr,
            attn, (long long)S * S, S, D, 0.03125f);

    // 3) softmax rows, in place (B*S rows of S)
    softmax_kernel<<<dim3(B * S), blk, 0, stream>>>(attn);

    // 4) Z = attn @ v  (batched, M=2048, N=1024, K=2048), A via TDM
    gemm_wmma_kernel<false, false, false, false, false>
        <<<dim3(D / 128, S / 128, B), blk, 0, stream>>>(
            attn, (long long)S * S, S, vb, (long long)S * D, D, nullptr,
            Zb, (long long)S * D, D, S, 1.0f);

    // 5) out = Z @ Wl_eff.T + bl  (M=8192, N=1024, K=1024, fp32 out), A+B via TDM
    gemm_wmma_kernel<false, false, true, true, true>
        <<<dim3(D / 128, (B * S) / 128, 1), blk, 0, stream>>>(
            Zb, 0, D, Weff, 0, D, bl, out, 0, D, D, 1.0f);
}
